// FullAttention_89807766159673
// MI455X (gfx1250) — compile-verified, matrix operand order checked by reading, not run on hardware
//
#include <hip/hip_runtime.h>
#include <hip/hip_bf16.h>

// ---------- problem constants ----------
#define B_   2
#define T_   2048
#define D_   1024
#define H_   16
#define KVH_ 4
#define HD_  64
#define GROUPS_ (H_ / KVH_)
#define M_  (B_ * T_)          // 4096 rows for all GEMMs
#define KVD_ (KVH_ * HD_)      // 256

typedef __bf16 bf16_t;
typedef __attribute__((ext_vector_type(16))) __bf16 v16bf;
typedef __attribute__((ext_vector_type(8)))  float  v8f;
typedef unsigned int u32x4 __attribute__((ext_vector_type(4)));
typedef int          i32x4 __attribute__((ext_vector_type(4)));
typedef int          i32x8 __attribute__((ext_vector_type(8)));

#if defined(__gfx1250__) && __has_builtin(__builtin_amdgcn_tensor_load_to_lds) && \
    __has_builtin(__builtin_amdgcn_s_wait_tensorcnt)
#define USE_TDM 1
#else
#define USE_TDM 0
#endif

// ---------- helpers ----------
__device__ __forceinline__ bf16_t f2bf(float x) {
    unsigned u = __builtin_bit_cast(unsigned, x);
    unsigned r = u + 0x7FFFu + ((u >> 16) & 1u);   // round-to-nearest-even
    unsigned short s = (unsigned short)(r >> 16);
    return __builtin_bit_cast(bf16_t, s);
}

struct U4Pair { uint4 lo; uint4 hi; };

// Load 16 bf16 (two 16-byte chunks) into the WMMA operand register layout:
// per-lane elements j=0..7 come from p, j=8..15 from p+16 (K = hi*8 + j and
// K = hi*8 + 16 + j per the 16-bit A/B VGPR layout).
__device__ __forceinline__ v16bf load16(const bf16_t* p) {
    U4Pair pr;
    pr.lo = *(const uint4*)(p);
    pr.hi = *(const uint4*)(p + 16);
    return __builtin_bit_cast(v16bf, pr);
}

__device__ __forceinline__ v8f wmma_bf16(v16bf a, v16bf b, v8f c) {
    return __builtin_amdgcn_wmma_f32_16x16x32_bf16(
        /*neg_a=*/false, a, /*neg_b=*/false, b,
        /*c_mod=*/(short)0, c, /*reuse_a=*/false, /*reuse_b=*/false);
}

__device__ __forceinline__ float rmax16(float x) {
#pragma unroll
    for (int m = 1; m < 16; m <<= 1) x = fmaxf(x, __shfl_xor(x, m, 32));
    return x;
}
__device__ __forceinline__ float rsum16(float x) {
#pragma unroll
    for (int m = 1; m < 16; m <<= 1) x += __shfl_xor(x, m, 32);
    return x;
}

#if USE_TDM
// Issue a TDM 2D-tile load: 64 rows x 64 bf16 columns from a row-major
// [rows, rowPitch] bf16 tensor into LDS (compact 64x64, 8 KB).
// D# layout per CDNA5 ISA 8.3-8.6 (group0: control/lds/global addr,
// group1: data_size, tensor dims/strides, tile dims; groups 2/3 unused=0).
__device__ __forceinline__ void tdm_load_tile_2d(unsigned lds_off,
                                                 const bf16_t* gptr,
                                                 unsigned tensor_d0,   // elements per row (K)
                                                 unsigned tensor_d1,   // rows (N)
                                                 unsigned tile_d0,     // 64
                                                 unsigned tile_d1) {   // 64
    unsigned long long ga = (unsigned long long)(uintptr_t)gptr;
    u32x4 g0;
    g0[0] = 1u;                                   // count=1, user desc
    g0[1] = lds_off;                              // lds_addr (bytes)
    g0[2] = (unsigned)(ga & 0xFFFFFFFFu);         // global_addr[31:0]
    g0[3] = (unsigned)((ga >> 32) & 0x01FFFFFFu)  // global_addr[56:32]
            | (2u << 30);                         // type=2 ("image")
    i32x8 g1;
    g1[0] = (int)(1u << 16);                      // data_size=1 (2B), wg_mask=0
    g1[1] = (int)((tensor_d0 & 0xFFFFu) << 16);   // tensor_dim0[15:0]
    g1[2] = (int)(((tensor_d0 >> 16) & 0xFFFFu) | ((tensor_d1 & 0xFFFFu) << 16));
    g1[3] = (int)(((tensor_d1 >> 16) & 0xFFFFu) | ((tile_d0 & 0xFFFFu) << 16));
    g1[4] = (int)(tile_d1 & 0xFFFFu);             // tile_dim1; tile_dim2=0
    g1[5] = (int)tensor_d0;                       // tensor_dim0_stride[31:0] = K
    g1[6] = 0;                                    // stride0 hi / stride1 lo
    g1[7] = 0;                                    // stride1 hi
    i32x4 z4 = {0, 0, 0, 0};
#if __clang_major__ >= 23
    i32x8 z8 = {0, 0, 0, 0, 0, 0, 0, 0};
    __builtin_amdgcn_tensor_load_to_lds(g0, g1, z4, z4, z8, 0);
#else
    __builtin_amdgcn_tensor_load_to_lds(g0, g1, z4, z4, 0);
#endif
}
#endif // USE_TDM

// ---------- elementwise conversion kernels ----------
__global__ void cvt_f32_bf16(const float* __restrict__ src,
                             bf16_t* __restrict__ dst, int n) {
    int i = blockIdx.x * blockDim.x + threadIdx.x;
    if (i < n) dst[i] = f2bf(src[i]);
}

// W [K,N] f32 -> Wt [N,K] bf16 (so GEMM B operand is contiguous along K)
__global__ void cvt_transpose_w(const float* __restrict__ W,
                                bf16_t* __restrict__ Wt, int K, int N) {
    int i = blockIdx.x * blockDim.x + threadIdx.x;
    if (i >= K * N) return;
    int k = i % K;
    int n = i / K;
    Wt[(size_t)n * K + k] = f2bf(W[(size_t)k * N + n]);
}

// ---------- bf16 WMMA GEMM:  C[M,N] (f32) = A[M,K](bf16) x Wt[N,K](bf16)^T
// block = 4 waves, block tile 64x64; each wave computes 16(M) x 64(N).
// Weight tile (shared by all 4 waves) staged in LDS by the Tensor Data Mover,
// double-buffered: TDM for chunk c+1 overlaps WMMAs on chunk c.
// All B fragments of a sub-step are loaded before the WMMA group so the
// ds reads pipeline (staggered dscnt waits) instead of load->wait->wmma chains.
__global__ void gemm_bf16(const bf16_t* __restrict__ A,
                          const bf16_t* __restrict__ Bt,
                          float* __restrict__ C,
                          int M, int N, int K) {
    __shared__ bf16_t ldsB[2][64 * 64];   // 2 x 8 KB chunks of Bt

    const int lane = threadIdx.x & 31;
    const int wave = threadIdx.x >> 5;
    const int ln = lane & 15;
    const int hi = lane >> 4;
    const int mBase = blockIdx.y * 64 + wave * 16;
    const int nBase = blockIdx.x * 64;
    const int nChunks = K >> 6;           // K / 64

    v8f acc[4] = {{}, {}, {}, {}};
    const bf16_t* arow = A + (size_t)(mBase + ln) * K;

#if USE_TDM
    const bf16_t* btile = Bt + (size_t)nBase * K;   // 64 rows starting at nBase
    if (wave == 0) {
        tdm_load_tile_2d((unsigned)(uintptr_t)&ldsB[0][0], btile,
                         (unsigned)K, (unsigned)N, 64u, 64u);
        __builtin_amdgcn_s_wait_tensorcnt(0);
    }
    __syncthreads();

    for (int c = 0; c < nChunks; ++c) {
        const int cur = c & 1;
        if (wave == 0 && (c + 1) < nChunks) {
            tdm_load_tile_2d((unsigned)(uintptr_t)&ldsB[cur ^ 1][0],
                             btile + (c + 1) * 64,
                             (unsigned)K, (unsigned)N, 64u, 64u);
        }
        __builtin_prefetch(arow + (c + 1) * 64, 0, 1);   // next A chunk

        // Issue ALL loads for this chunk first (2 A frags + 8 B frags) ...
        v16bf a0 = load16(arow + c * 64 + hi * 8);
        v16bf a1 = load16(arow + c * 64 + 32 + hi * 8);
        v16bf b0[4], b1[4];
#pragma unroll
        for (int nt = 0; nt < 4; ++nt)
            b0[nt] = load16(&ldsB[cur][(nt * 16 + ln) * 64 + hi * 8]);
#pragma unroll
        for (int nt = 0; nt < 4; ++nt)
            b1[nt] = load16(&ldsB[cur][(nt * 16 + ln) * 64 + 32 + hi * 8]);
        // ... then run the 8 WMMAs back-to-back.
#pragma unroll
        for (int nt = 0; nt < 4; ++nt)
            acc[nt] = wmma_bf16(a0, b0[nt], acc[nt]);
#pragma unroll
        for (int nt = 0; nt < 4; ++nt)
            acc[nt] = wmma_bf16(a1, b1[nt], acc[nt]);

        if (wave == 0 && (c + 1) < nChunks)
            __builtin_amdgcn_s_wait_tensorcnt(0);
        __syncthreads();
    }
#else
    for (int k0 = 0; k0 < K; k0 += 32) {
        const int kb = k0 + hi * 8;
        v16bf a = load16(arow + kb);
        v16bf b[4];
#pragma unroll
        for (int nt = 0; nt < 4; ++nt)
            b[nt] = load16(Bt + (size_t)(nBase + nt * 16 + ln) * K + kb);
#pragma unroll
        for (int nt = 0; nt < 4; ++nt)
            acc[nt] = wmma_bf16(a, b[nt], acc[nt]);
    }
#endif

#pragma unroll
    for (int nt = 0; nt < 4; ++nt)
#pragma unroll
        for (int r = 0; r < 8; ++r) {
            int rr = mBase + r + 8 * hi;
            int cc = nBase + nt * 16 + ln;
            C[(size_t)rr * N + cc] = acc[nt][r];
        }
}

// ---------- RoPE + transpose: src f32 [B,T,heads,64] -> dst bf16 [B,heads,T,64]
__global__ void rope_transpose(const float* __restrict__ src,
                               const float* __restrict__ cosT,
                               const float* __restrict__ sinT,
                               bf16_t* __restrict__ dst, int heads) {
    int i = blockIdx.x * blockDim.x + threadIdx.x;
    int total = B_ * T_ * heads * HD_;
    if (i >= total) return;
    int d = i & 63;
    int tmp = i >> 6;
    int hh = tmp % heads;
    int tmp2 = tmp / heads;
    int t = tmp2 % T_;
    int bb = tmp2 / T_;
    float x = src[i];
    int dp = (d < 32) ? d + 32 : d - 32;
    float xo = src[i - d + dp];
    float c = cosT[t * 32 + (d & 31)];
    float s = sinT[t * 32 + (d & 31)];
    float rot = (d < 32) ? -xo : xo;
    dst[(((size_t)bb * heads + hh) * T_ + t) * HD_ + d] = f2bf(x * c + rot * s);
}

// V: src f32 [B,T,KVH,64] -> Vt bf16 [B,KVH,64,T]  (contiguous along T for PV WMMA)
__global__ void v_transpose(const float* __restrict__ src,
                            bf16_t* __restrict__ dst) {
    int i = blockIdx.x * blockDim.x + threadIdx.x;
    int total = B_ * T_ * KVH_ * HD_;
    if (i >= total) return;
    int d = i & 63;
    int tmp = i >> 6;
    int hh = tmp % KVH_;
    int tmp2 = tmp / KVH_;
    int t = tmp2 % T_;
    int bb = tmp2 / T_;
    dst[(((size_t)bb * KVH_ + hh) * HD_ + d) * T_ + t] = f2bf(src[i]);
}

// ---------- flash attention: one wave per 16-query tile ----------
// Qt [B,H,T,64] bf16 (rope'd), Kt [B,KVH,T,64] bf16 (rope'd), Vt [B,KVH,64,T] bf16
// Out: attnO bf16 [B,T,H*64] row-major (GEMM A operand for the O projection).
__global__ void attn_kernel(const bf16_t* __restrict__ Qt,
                            const bf16_t* __restrict__ Kt,
                            const bf16_t* __restrict__ Vt,
                            bf16_t* __restrict__ attnO) {
    __shared__ bf16_t lds_p[16 * 32];

    const int lane = threadIdx.x & 31;
    const int ln = lane & 15;
    const int hi = lane >> 4;
    const int qtile = blockIdx.x;
    const int h = blockIdx.y;
    const int b = blockIdx.z;
    const int kvh = h / GROUPS_;
    const int q0 = qtile * 16;

    // Q tile in A layout: two regs covering head-dim K=0..31 and 32..63
    const bf16_t* qrow = Qt + (((size_t)b * H_ + h) * T_ + q0 + ln) * HD_;
    v16bf qa0 = load16(qrow + hi * 8);
    v16bf qa1 = load16(qrow + 32 + hi * 8);

    float m_r[8], l_r[8];
#pragma unroll
    for (int r = 0; r < 8; ++r) { m_r[r] = -1e30f; l_r[r] = 0.0f; }
    v8f o[4] = {{}, {}, {}, {}};
    const v8f zero = {};

    const bf16_t* kbase = Kt + ((size_t)b * KVH_ + kvh) * T_ * HD_;
    const bf16_t* vbase = Vt + ((size_t)b * KVH_ + kvh) * HD_ * T_;
    const float scale = 0.125f;  // 1/sqrt(64)

    for (int kb0 = 0; kb0 < q0 + 16; kb0 += 32) {
        // ---- issue all 4 K fragment loads, then both S-tile WMMA chains
        v16bf kf[2][2];
#pragma unroll
        for (int st = 0; st < 2; ++st) {
            const bf16_t* krow = kbase + (size_t)(kb0 + st * 16 + ln) * HD_;
            kf[st][0] = load16(krow + hi * 8);
            kf[st][1] = load16(krow + 32 + hi * 8);
        }
        v8f s[2];
#pragma unroll
        for (int st = 0; st < 2; ++st)
            s[st] = wmma_bf16(qa0, kf[st][0], wmma_bf16(qa1, kf[st][1], zero));

        // ---- online softmax (rows live across the 16-lane half holding one VGPR)
#pragma unroll
        for (int r = 0; r < 8; ++r) {
            int q_idx = q0 + r + 8 * hi;
            float v0 = s[0][r] * scale;
            float v1 = s[1][r] * scale;
            if (kb0 + ln > q_idx) v0 = -1e30f;
            if (kb0 + 16 + ln > q_idx) v1 = -1e30f;
            float rowm = fmaxf(rmax16(v0), rmax16(v1));
            float m_new = fmaxf(m_r[r], rowm);
            float alpha = __expf(m_r[r] - m_new);
            float p0 = __expf(v0 - m_new);
            float p1 = __expf(v1 - m_new);
            l_r[r] = l_r[r] * alpha + rsum16(p0) + rsum16(p1);
            m_r[r] = m_new;
            lds_p[(r + 8 * hi) * 32 + ln] = f2bf(p0);
            lds_p[(r + 8 * hi) * 32 + 16 + ln] = f2bf(p1);
#pragma unroll
            for (int nt = 0; nt < 4; ++nt) o[nt][r] *= alpha;
        }
        __syncthreads();
        // ---- reload P (16x32) in A layout from LDS
        v16bf pa = load16(&lds_p[ln * 32 + hi * 8]);
        __syncthreads();
        // ---- O += P V : issue all 4 V fragment loads, then 4 WMMAs
        v16bf vfrag[4];
#pragma unroll
        for (int nt = 0; nt < 4; ++nt)
            vfrag[nt] = load16(vbase + (size_t)(nt * 16 + ln) * T_ + kb0 + hi * 8);
#pragma unroll
        for (int nt = 0; nt < 4; ++nt)
            o[nt] = wmma_bf16(pa, vfrag[nt], o[nt]);
    }

    // ---- finalize & store [B,T,H*64]
#pragma unroll
    for (int nt = 0; nt < 4; ++nt)
#pragma unroll
        for (int r = 0; r < 8; ++r) {
            int t = q0 + r + 8 * hi;
            float inv = 1.0f / l_r[r];
            attnO[((size_t)b * T_ + t) * D_ + h * HD_ + nt * 16 + ln] =
                f2bf(o[nt][r] * inv);
        }
}

// ---------- host side ----------
extern "C" void kernel_launch(void* const* d_in, const int* in_sizes, int n_in,
                              void* d_out, int out_size, void* d_ws, size_t ws_size,
                              hipStream_t stream) {
    const float* x        = (const float*)d_in[0];
    const float* rope_cos = (const float*)d_in[1];
    const float* rope_sin = (const float*)d_in[2];
    // d_in[3] = attention_mask (causal; implemented directly in the kernel)
    const float* Wq = (const float*)d_in[4];
    const float* Wk = (const float*)d_in[5];
    const float* Wv = (const float*)d_in[6];
    const float* Wo = (const float*)d_in[7];
    float* out = (float*)d_out;

    char* w = (char*)d_ws;
    auto alloc = [&](size_t bytes) -> void* {
        void* p = (void*)w;
        w += (bytes + 255) & ~(size_t)255;
        return p;
    };
    bf16_t* xb    = (bf16_t*)alloc((size_t)M_ * D_ * 2);
    bf16_t* wq_t  = (bf16_t*)alloc((size_t)D_ * D_ * 2);
    bf16_t* wk_t  = (bf16_t*)alloc((size_t)KVD_ * D_ * 2);
    bf16_t* wv_t  = (bf16_t*)alloc((size_t)KVD_ * D_ * 2);
    bf16_t* wo_t  = (bf16_t*)alloc((size_t)D_ * D_ * 2);
    float*  qf    = (float*)alloc((size_t)M_ * D_ * 4);
    float*  kf    = (float*)alloc((size_t)M_ * KVD_ * 4);
    float*  vf    = (float*)alloc((size_t)M_ * KVD_ * 4);
    bf16_t* Qt    = (bf16_t*)alloc((size_t)B_ * H_ * T_ * HD_ * 2);
    bf16_t* Kt    = (bf16_t*)alloc((size_t)B_ * KVH_ * T_ * HD_ * 2);
    bf16_t* Vt    = (bf16_t*)alloc((size_t)B_ * KVH_ * HD_ * T_ * 2);
    bf16_t* attnO = (bf16_t*)alloc((size_t)M_ * D_ * 2);

    const int TPB = 256;
    // 1) convert activations & weights to bf16 (weights transposed to [N,K])
    cvt_f32_bf16<<<(M_ * D_ + TPB - 1) / TPB, TPB, 0, stream>>>(x, xb, M_ * D_);
    cvt_transpose_w<<<(D_ * D_ + TPB - 1) / TPB, TPB, 0, stream>>>(Wq, wq_t, D_, D_);
    cvt_transpose_w<<<(D_ * KVD_ + TPB - 1) / TPB, TPB, 0, stream>>>(Wk, wk_t, D_, KVD_);
    cvt_transpose_w<<<(D_ * KVD_ + TPB - 1) / TPB, TPB, 0, stream>>>(Wv, wv_t, D_, KVD_);
    cvt_transpose_w<<<(D_ * D_ + TPB - 1) / TPB, TPB, 0, stream>>>(Wo, wo_t, D_, D_);

    // 2) QKV projections (WMMA GEMMs, TDM-staged weight tiles)
    gemm_bf16<<<dim3(D_ / 64, M_ / 64), 128, 0, stream>>>(xb, wq_t, qf, M_, D_, D_);
    gemm_bf16<<<dim3(KVD_ / 64, M_ / 64), 128, 0, stream>>>(xb, wk_t, kf, M_, KVD_, D_);
    gemm_bf16<<<dim3(KVD_ / 64, M_ / 64), 128, 0, stream>>>(xb, wv_t, vf, M_, KVD_, D_);

    // 3) RoPE + head transpose
    rope_transpose<<<(M_ * D_ + TPB - 1) / TPB, TPB, 0, stream>>>(qf, rope_cos, rope_sin, Qt, H_);
    rope_transpose<<<(M_ * KVD_ + TPB - 1) / TPB, TPB, 0, stream>>>(kf, rope_cos, rope_sin, Kt, KVH_);
    v_transpose<<<(M_ * KVD_ + TPB - 1) / TPB, TPB, 0, stream>>>(vf, Vt);

    // 4) flash attention, one wave32 per 16-query tile
    attn_kernel<<<dim3(T_ / 16, H_, B_), 32, 0, stream>>>(Qt, Kt, Vt, attnO);

    // 5) output projection -> fp32 result
    gemm_bf16<<<dim3(D_ / 64, M_ / 64), 128, 0, stream>>>(attnO, wo_t, out, M_, D_, D_);
}
